// Loss_forward_45552423141441
// MI455X (gfx1250) — matrix operand order repaired and checked by reading, not compiled, not run on hardware
//
#include <hip/hip_runtime.h>
#include <hip/hip_bf16.h>
#include <math.h>

// ---------------------------------------------------------------------------
// Problem constants (from reference)
// ---------------------------------------------------------------------------
#define IMG     384
#define CH      64
#define PATCH   7
#define PP      49          // PATCH*PATCH
#define STRIDE  3
#define GRID    126         // (384-7)/3+1
#define S       4096        // SAMPLE*SAMPLE
#define KDIM    3136        // CH*PP  (GEMM K)
#define TILE    128
#define KSTEP   32
#define KITERS  98          // KDIM / KSTEP

typedef __attribute__((ext_vector_type(16))) _Float16 v16h;
typedef __attribute__((ext_vector_type(8)))  float    v8f;
typedef __attribute__((vector_size(16)))     int      v4i;   // matches builtin proto

#define AS1 __attribute__((address_space(1)))
#define AS3 __attribute__((address_space(3)))

// CDNA5 async global->LDS copy path (ASYNCcnt-tracked), guarded so the file
// still compiles on toolchains lacking the builtins.
#if __has_builtin(__builtin_amdgcn_global_load_async_to_lds_b128)
#define HAVE_ASYNC_LDS 1
#if __has_builtin(__builtin_amdgcn_s_wait_asynccnt)
#define WAIT_ASYNC(n) __builtin_amdgcn_s_wait_asynccnt(n)
#else
#define WAIT_ASYNC(n) asm volatile("s_wait_asynccnt %0" ::"i"(n) : "memory")
#endif
#else
#define HAVE_ASYNC_LDS 0
#define WAIT_ASYNC(n) ((void)0)
#endif

// ---------------------------------------------------------------------------
// K_init: zero accumulators, set argmin keys to +inf
// ---------------------------------------------------------------------------
__global__ __launch_bounds__(256) void k_init(float* nX, float* nY, float* counts,
                                              unsigned long long* minkey, float* out) {
    int i = blockIdx.x * 256 + threadIdx.x;
    if (i < S) {
        nX[i] = 0.f; nY[i] = 0.f; counts[i] = 0.f;
        minkey[i] = ~0ull;
    }
    if (i == 0) out[0] = 0.f;
}

// ---------------------------------------------------------------------------
// K_field: grid_sample(nearest, align_corners=False) index precompute
// ---------------------------------------------------------------------------
__global__ __launch_bounds__(256) void k_field(const float* __restrict__ tf,
                                               const float* __restrict__ rf,
                                               int* ixt, int* iyt, int* ixr, int* iyr) {
    int i = blockIdx.x * 256 + threadIdx.x;
    if (i >= S) return;
    // W = H = 126 -> (g+1)*W/2 - 0.5 = (g+1)*63 - 0.5 ; round-half-even like jnp.round
    float gx = tf[2 * i], gy = tf[2 * i + 1];
    ixt[i] = min(GRID - 1, max(0, (int)rintf((gx + 1.f) * 63.f - 0.5f)));
    iyt[i] = min(GRID - 1, max(0, (int)rintf((gy + 1.f) * 63.f - 0.5f)));
    gx = rf[2 * i]; gy = rf[2 * i + 1];
    ixr[i] = min(GRID - 1, max(0, (int)rintf((gx + 1.f) * 63.f - 0.5f)));
    iyr[i] = min(GRID - 1, max(0, (int)rintf((gy + 1.f) * 63.f - 0.5f)));
}

// ---------------------------------------------------------------------------
// K_mean: per unfold-row mean of the sampled reference features
// (reference centers BOTH operands with the y-mean)
// ---------------------------------------------------------------------------
__global__ __launch_bounds__(256) void k_mean(const float* __restrict__ fr,
                                              const int* __restrict__ ixr,
                                              const int* __restrict__ iyr,
                                              float* __restrict__ meanY) {
    __shared__ float red[256];
    const int r = blockIdx.x;               // 0..3135
    const int c = r / PP, pq = r % PP, ki = pq / PATCH, kj = pq % PATCH;
    const int tid = threadIdx.x;
    const float* base = fr + (size_t)c * IMG * IMG;
    float sum = 0.f;
    for (int u = 0; u < 16; ++u) {
        int t = u * 256 + tid;
        sum += base[(size_t)(iyr[t] * STRIDE + ki) * IMG + ixr[t] * STRIDE + kj];
    }
    red[tid] = sum;
    __syncthreads();
    for (int off = 128; off; off >>= 1) {
        if (tid < off) red[tid] += red[tid + off];
        __syncthreads();
    }
    if (tid == 0) meanY[r] = red[0] * (1.f / (float)S);
}

// ---------------------------------------------------------------------------
// K_write: gather + center + f16-pack DIRECTLY into GEMM-ready tiled layout:
//   xcP[(kb*S + s)*32 + (k%32)],  kb = k/32
// Each 128x32 GEMM LDS tile is then an 8KB CONTIGUOUS span -> async-copyable.
// Also accumulates per-sample squared channel norms (98 partials per sample).
// ---------------------------------------------------------------------------
__global__ __launch_bounds__(256) void k_write(const float* __restrict__ ft,
                                               const float* __restrict__ fr,
                                               const int* __restrict__ ixt, const int* __restrict__ iyt,
                                               const int* __restrict__ ixr, const int* __restrict__ iyr,
                                               const float* __restrict__ meanY,
                                               _Float16* __restrict__ xcP, _Float16* __restrict__ ycP,
                                               float* __restrict__ nX, float* __restrict__ nY) {
    const int idx = blockIdx.x * 256 + threadIdx.x;   // kb*S + s, total 98*4096
    const int kb = idx >> 12, s = idx & (S - 1);
    const int xt = ixt[s] * STRIDE, yt = iyt[s] * STRIDE;
    const int xr = ixr[s] * STRIDE, yr = iyr[s] * STRIDE;

    union { _Float16 h[32]; uint4 v[4]; } tx, ty;
    float sx = 0.f, sy = 0.f;
#pragma unroll
    for (int kk = 0; kk < 32; ++kk) {
        int r = kb * 32 + kk;
        int c = r / PP, pq = r % PP, ki = pq / PATCH, kj = pq % PATCH;
        float m = meanY[r];
        float yv = fr[((size_t)c * IMG + yr + ki) * IMG + xr + kj] - m;
        float xv = ft[((size_t)c * IMG + yt + ki) * IMG + xt + kj] - m;
        ty.h[kk] = (_Float16)yv;
        tx.h[kk] = (_Float16)xv;
        sy += yv * yv;
        sx += xv * xv;
    }
    uint4* dx = (uint4*)(xcP + (size_t)idx * 32);
    uint4* dy = (uint4*)(ycP + (size_t)idx * 32);
#pragma unroll
    for (int q = 0; q < 4; ++q) { dx[q] = tx.v[q]; dy[q] = ty.v[q]; }
    atomicAdd(&nX[s], sx);
    atomicAdd(&nY[s], sy);
}

// ---------------------------------------------------------------------------
// K_orient_gather: 98 blocks, one per (channel c in 0..1, patch i in 0..48)
// layout: ox[(c*49 + i)*S + s]  (f32, 1.6MB each -> L2 resident)
// ---------------------------------------------------------------------------
__global__ __launch_bounds__(256) void k_orient_gather(const float* __restrict__ ot,
                                                       const float* __restrict__ orr,
                                                       const int* __restrict__ ixt, const int* __restrict__ iyt,
                                                       const int* __restrict__ ixr, const int* __restrict__ iyr,
                                                       float* __restrict__ ox, float* __restrict__ oy) {
    const int b = blockIdx.x;               // 0..97
    const int c = b / PP, i = b % PP, ki = i / PATCH, kj = i % PATCH;
    const int tid = threadIdx.x;
    const float* bt = ot + (size_t)c * IMG * IMG;
    const float* br = orr + (size_t)c * IMG * IMG;
    for (int u = 0; u < 16; ++u) {
        int t = u * 256 + tid;
        ox[(size_t)b * S + t] = bt[(size_t)(iyt[t] * STRIDE + ki) * IMG + ixt[t] * STRIDE + kj];
        oy[(size_t)b * S + t] = br[(size_t)(iyr[t] * STRIDE + ki) * IMG + ixr[t] * STRIDE + kj];
    }
}

// ---------------------------------------------------------------------------
// K_orient_norm: XN[i][s] = |x_i(s)|^2  (2-channel), same for YN
// ---------------------------------------------------------------------------
__global__ __launch_bounds__(256) void k_orient_norm(const float* __restrict__ ox,
                                                     const float* __restrict__ oy,
                                                     float* __restrict__ XN, float* __restrict__ YN) {
    int idx = blockIdx.x * 256 + threadIdx.x;
    if (idx >= PP * S) return;
    float a0 = ox[idx], a1 = ox[(size_t)PP * S + idx];
    XN[idx] = a0 * a0 + a1 * a1;
    float b0 = oy[idx], b1 = oy[(size_t)PP * S + idx];
    YN[idx] = b0 * b0 + b1 * b1;
}

// ---------------------------------------------------------------------------
// K_invnorm: turn sum-of-squares into 1/(||.||+eps), in place
// ---------------------------------------------------------------------------
__global__ __launch_bounds__(256) void k_invnorm(float* nX, float* nY) {
    int i = blockIdx.x * 256 + threadIdx.x;
    if (i >= S) return;
    nX[i] = 1.f / (sqrtf(nX[i]) + 1e-5f);
    nY[i] = 1.f / (sqrtf(nY[i]) + 1e-5f);
}

// ---------------------------------------------------------------------------
// K_gemm: D[s,t] = clip((1-cos)/2) + 0.5*orient_d, fused epilogue + row argmin.
// 128x128 tile per 256-thread block (8 waves; wave = 64M x 32N = 4x2 WMMA tiles)
// v_wmma_f32_16x16x32_f16; operands pre-packed so each 8KB tile is contiguous
// and double-buffered via GLOBAL_LOAD_ASYNC_TO_LDS_B128 (ASYNCcnt) when
// available, plain b128 copy otherwise.
// ---------------------------------------------------------------------------
__global__ __launch_bounds__(256) void k_gemm(const _Float16* __restrict__ xcP,
                                              const _Float16* __restrict__ ycP,
                                              const float* __restrict__ invX,
                                              const float* __restrict__ invY,
                                              const float* __restrict__ ox,
                                              const float* __restrict__ oy,
                                              const float* __restrict__ XN,
                                              const float* __restrict__ YN,
                                              float* __restrict__ D,
                                              unsigned long long* __restrict__ minkey) {
    __shared__ __align__(16) _Float16 As[2][TILE][KSTEP];   // [m][k] halves, 8KB/buffer
    __shared__ __align__(16) _Float16 Bs[2][TILE][KSTEP];   // [n][k]
    __shared__ unsigned long long rowkey[TILE];

    const int tid  = threadIdx.x;
    const int lane = tid & 31, wave = tid >> 5;
    const int wm = wave & 1, wn = wave >> 1;       // 2x4 wave grid
    const int half = lane >> 4, lr = lane & 15;
    const int sBlock = blockIdx.x * TILE, tBlock = blockIdx.y * TILE;

    // Contiguous tile copy: tile kb for rows [sBlock,+128) lives at
    // xcP + (kb*S + sBlock)*32 halves, 8192 bytes. 32 bytes per thread.
    auto copy_tile = [&](int step, int buf) {
        const char* gA = (const char*)(xcP + (((size_t)step * S + sBlock) << 5)) + (tid << 5);
        const char* gB = (const char*)(ycP + (((size_t)step * S + tBlock) << 5)) + (tid << 5);
        char* lA = (char*)&As[buf][0][0] + (tid << 5);
        char* lB = (char*)&Bs[buf][0][0] + (tid << 5);
#if HAVE_ASYNC_LDS
        __builtin_amdgcn_global_load_async_to_lds_b128((AS1 v4i*)gA, (AS3 v4i*)lA, 0, 0);
        __builtin_amdgcn_global_load_async_to_lds_b128((AS1 v4i*)gA, (AS3 v4i*)lA, 16, 0);
        __builtin_amdgcn_global_load_async_to_lds_b128((AS1 v4i*)gB, (AS3 v4i*)lB, 0, 0);
        __builtin_amdgcn_global_load_async_to_lds_b128((AS1 v4i*)gB, (AS3 v4i*)lB, 16, 0);
#else
        ((uint4*)lA)[0] = ((const uint4*)gA)[0];
        ((uint4*)lA)[1] = ((const uint4*)gA)[1];
        ((uint4*)lB)[0] = ((const uint4*)gB)[0];
        ((uint4*)lB)[1] = ((const uint4*)gB)[1];
#endif
    };

    v8f acc[4][2];
#pragma unroll
    for (int mt = 0; mt < 4; ++mt)
#pragma unroll
        for (int nt = 0; nt < 2; ++nt) acc[mt][nt] = (v8f)0.f;

    auto compute = [&](int buf) {
        union Frag { unsigned u[8]; v16h h; };
        Frag af[4], bf[2];
#pragma unroll
        for (int mt = 0; mt < 4; ++mt) {
            // A 16x32 f16 layout: lane-half h holds K={2v,2v+1}+8h (v<4), 16+{..}+8h (v>=4)
            const unsigned* rp = (const unsigned*)&As[buf][wm * 64 + mt * 16 + lr][0];
#pragma unroll
            for (int v = 0; v < 4; ++v) af[mt].u[v] = rp[v + 4 * half];
#pragma unroll
            for (int v = 4; v < 8; ++v) af[mt].u[v] = rp[8 + (v - 4) + 4 * half];
        }
#pragma unroll
        for (int nt = 0; nt < 2; ++nt) {
            // B 32x16 f16 layout: lane-half h holds K={2v,2v+1}+8h (v<4), 16+{..}+8h (v>=4)
            const unsigned* rp = (const unsigned*)&Bs[buf][wn * 32 + nt * 16 + lr][0];
#pragma unroll
            for (int v = 0; v < 8; ++v) bf[nt].u[v] = rp[(v & 3) + 4 * half + 8 * (v >> 2)];
        }
#pragma unroll
        for (int mt = 0; mt < 4; ++mt)
#pragma unroll
            for (int nt = 0; nt < 2; ++nt)
                acc[mt][nt] = __builtin_amdgcn_wmma_f32_16x16x32_f16(
                    false, af[mt].h, false, bf[nt].h, (short)0, acc[mt][nt], false, false);
    };

    // depth-2 async pipeline: tiles {k,k+1} in flight; 4 loads/lane per tile,
    // in-order completion => asynccnt<=4 means tile k has landed.
    copy_tile(0, 0);
    copy_tile(1, 1);
    for (int step = 0; step < KITERS; ++step) {
        if (step + 1 < KITERS) { WAIT_ASYNC(4); } else { WAIT_ASYNC(0); }
        __syncthreads();
        compute(step & 1);
        __syncthreads();
        if (step + 2 < KITERS) copy_tile(step + 2, step & 1);
    }

    // ---- fused epilogue: cosine normalize + clip, then orient distance -----
    // C/D layout: VGPR j -> M = j + 8*(lane/16), N = lane%16
#pragma unroll
    for (int mt = 0; mt < 4; ++mt)
#pragma unroll
        for (int j = 0; j < 8; ++j) {
            int s = sBlock + wm * 64 + mt * 16 + 8 * half + j;
            float ixn = invX[s];
#pragma unroll
            for (int nt = 0; nt < 2; ++nt) {
                int t = tBlock + wn * 32 + nt * 16 + lr;
                float cosv = acc[mt][nt][j] * ixn * invY[t];
                acc[mt][nt][j] = fmaxf(0.f, (1.f - cosv) * 0.5f);
            }
        }

    // orient: d += ORIENT_W * (1/49) * min(clip(xn-2dot+yn),clip(xn+2dot+yn))/2
    const float OSC = 0.5f * 0.5f / (float)PP;   // ORIENT_W * (1/2) * (1/pp)
    for (int i = 0; i < PP; ++i) {
        float y0[2], y1[2], yn[2];
#pragma unroll
        for (int nt = 0; nt < 2; ++nt) {
            int t = tBlock + wn * 32 + nt * 16 + lr;
            y0[nt] = oy[(size_t)i * S + t];
            y1[nt] = oy[(size_t)(PP + i) * S + t];
            yn[nt] = YN[(size_t)i * S + t];
        }
#pragma unroll
        for (int mt = 0; mt < 4; ++mt)
#pragma unroll
            for (int j = 0; j < 8; ++j) {
                int s = sBlock + wm * 64 + mt * 16 + 8 * half + j;   // broadcast loads
                float x0 = ox[(size_t)i * S + s];
                float x1 = ox[(size_t)(PP + i) * S + s];
                float xn = XN[(size_t)i * S + s];
#pragma unroll
                for (int nt = 0; nt < 2; ++nt) {
                    float dot = fmaf(x0, y0[nt], x1 * y1[nt]);
                    float sxy = xn + yn[nt];
                    float d1 = fmaxf(0.f, sxy - 2.f * dot);
                    float d2 = fmaxf(0.f, sxy + 2.f * dot);
                    acc[mt][nt][j] = fmaf(fminf(d1, d2), OSC, acc[mt][nt][j]);
                }
            }
    }

    // ---- store D and per-row argmin (packed u64 keys, d>=0 => monotone) ----
    if (tid < TILE) rowkey[tid] = ~0ull;
    __syncthreads();
#pragma unroll
    for (int mt = 0; mt < 4; ++mt)
#pragma unroll
        for (int j = 0; j < 8; ++j) {
            int sl = wm * 64 + mt * 16 + 8 * half + j;
            int s = sBlock + sl;
            unsigned long long best = ~0ull;
#pragma unroll
            for (int nt = 0; nt < 2; ++nt) {
                int t = tBlock + wn * 32 + nt * 16 + lr;
                float dv = acc[mt][nt][j];
                D[(size_t)s * S + t] = dv;
                unsigned long long key =
                    ((unsigned long long)__float_as_uint(dv) << 32) | (unsigned)t;
                best = key < best ? key : best;
            }
            atomicMin(&rowkey[sl], best);
        }
    __syncthreads();
    if (tid < TILE) atomicMin(&minkey[sBlock + tid], rowkey[tid]);
}

// ---------------------------------------------------------------------------
// K_counts: histogram of per-row argmin columns (batch 0 only, N=1)
// ---------------------------------------------------------------------------
__global__ __launch_bounds__(256) void k_counts(const unsigned long long* __restrict__ minkey,
                                                float* __restrict__ counts) {
    int s = blockIdx.x * 256 + threadIdx.x;
    if (s >= S) return;
    int k = (int)(minkey[s] & 0xFFFFFFFFull);
    atomicAdd(&counts[k], 1.0f);
}

// ---------------------------------------------------------------------------
// K_loss: per-row min + log-sum-exp over D + penalty ; mean via atomicAdd.
//  -log(max nw) = log(sum_t exp((1-rel_t)/H)) - (1-rel_min)/H
// ---------------------------------------------------------------------------
__global__ __launch_bounds__(256) void k_loss(const float* __restrict__ D,
                                              const float* __restrict__ counts,
                                              float* __restrict__ out) {
    __shared__ float red[256];
    const int s = blockIdx.x, tid = threadIdx.x;
    float v[16];
    float mn = 3.4e38f;
#pragma unroll
    for (int q = 0; q < 16; ++q) {
        int t = q * 256 + tid;
        // norm_factor = S/K = 1 here; penalty = counts * OCC_W
        v[q] = D[(size_t)s * S + t] + counts[t] * 0.05f;
        mn = fminf(mn, v[q]);
    }
    red[tid] = mn;
    __syncthreads();
    for (int off = 128; off; off >>= 1) {
        if (tid < off) red[tid] = fminf(red[tid], red[tid + off]);
        __syncthreads();
    }
    const float dmin = red[0];
    __syncthreads();
    const float inv = 1.f / (dmin + 1e-5f);
    float se = 0.f;
#pragma unroll
    for (int q = 0; q < 16; ++q)
        se += __expf((1.f - v[q] * inv) * 2.f);    // 1/H_PARAM = 2
    red[tid] = se;
    __syncthreads();
    for (int off = 128; off; off >>= 1) {
        if (tid < off) red[tid] += red[tid + off];
        __syncthreads();
    }
    if (tid == 0) {
        float lossrow = logf(red[0]) - (1.f - dmin * inv) * 2.f;
        atomicAdd(out, lossrow * (1.f / (float)S));
    }
}

// ---------------------------------------------------------------------------
// Launcher
// ---------------------------------------------------------------------------
extern "C" void kernel_launch(void* const* d_in, const int* in_sizes, int n_in,
                              void* d_out, int out_size, void* d_ws, size_t ws_size,
                              hipStream_t stream) {
    const float* tfeat  = (const float*)d_in[0];
    const float* rfeat  = (const float*)d_in[1];
    const float* torien = (const float*)d_in[2];
    const float* rorien = (const float*)d_in[3];
    const float* tfield = (const float*)d_in[4];
    const float* rfield = (const float*)d_in[5];
    float* out = (float*)d_out;

    char* ws = (char*)d_ws;
    size_t off = 0;
    auto alloc = [&](size_t bytes) -> void* {
        void* p = ws + off;
        off = (off + bytes + 255) & ~(size_t)255;
        return p;
    };
    _Float16* xcP = (_Float16*)alloc((size_t)KDIM * S * 2);   // packed, tile-contiguous
    _Float16* ycP = (_Float16*)alloc((size_t)KDIM * S * 2);
    float* ox = (float*)alloc((size_t)2 * PP * S * 4);
    float* oy = (float*)alloc((size_t)2 * PP * S * 4);
    float* XN = (float*)alloc((size_t)PP * S * 4);
    float* YN = (float*)alloc((size_t)PP * S * 4);
    float* nX = (float*)alloc((size_t)S * 4);
    float* nY = (float*)alloc((size_t)S * 4);
    float* meanY = (float*)alloc((size_t)KDIM * 4);
    int* ixt = (int*)alloc((size_t)S * 4);
    int* iyt = (int*)alloc((size_t)S * 4);
    int* ixr = (int*)alloc((size_t)S * 4);
    int* iyr = (int*)alloc((size_t)S * 4);
    unsigned long long* minkey = (unsigned long long*)alloc((size_t)S * 8);
    float* counts = (float*)alloc((size_t)S * 4);
    float* D = (float*)alloc((size_t)S * S * 4);

    k_init<<<16, 256, 0, stream>>>(nX, nY, counts, minkey, out);
    k_field<<<16, 256, 0, stream>>>(tfield, rfield, ixt, iyt, ixr, iyr);
    k_mean<<<KDIM, 256, 0, stream>>>(rfeat, ixr, iyr, meanY);
    k_write<<<(KITERS * S) / 256, 256, 0, stream>>>(tfeat, rfeat, ixt, iyt, ixr, iyr,
                                                    meanY, xcP, ycP, nX, nY);
    k_orient_gather<<<2 * PP, 256, 0, stream>>>(torien, rorien, ixt, iyt, ixr, iyr, ox, oy);
    k_orient_norm<<<(PP * S + 255) / 256, 256, 0, stream>>>(ox, oy, XN, YN);
    k_invnorm<<<16, 256, 0, stream>>>(nX, nY);
    k_gemm<<<dim3(S / TILE, S / TILE), 256, 0, stream>>>(xcP, ycP, nX, nY, ox, oy, XN, YN, D, minkey);
    k_counts<<<16, 256, 0, stream>>>(minkey, counts);
    k_loss<<<S, 256, 0, stream>>>(D, counts, out);
}
